// PointNet_89438398972568
// MI455X (gfx1250) — compile-verified, matrix operand order checked by reading, not run on hardware
//
#include <hip/hip_runtime.h>

#define NEGV (-1e30f)

typedef __attribute__((ext_vector_type(16))) __bf16 v16bf;
typedef __attribute__((ext_vector_type(8)))  float  v8f;
typedef __attribute__((vector_size(16)))     int    vi4;   // b128 payload type

// ---------------------------------------------------------------------------
// CDNA5 async global->LDS copy (ASYNCcnt-tracked), with generic-copy fallback.
// Builtin signature (from hipcc diagnostic): (vi4 AS1*, vi4 AS3*, int, int)
// ---------------------------------------------------------------------------
#if defined(__gfx1250__) && __has_builtin(__builtin_amdgcn_global_load_async_to_lds_b128)
#define ASYNC_LDS 1
#endif

__device__ __forceinline__ void copy16(__bf16* dst_lds, const __bf16* src_g) {
#if ASYNC_LDS
  auto g = (__attribute__((address_space(1))) vi4*)(unsigned long long)(uintptr_t)src_g;
  auto l = (__attribute__((address_space(3))) vi4*)(unsigned)(uintptr_t)dst_lds;
  __builtin_amdgcn_global_load_async_to_lds_b128(g, l, 0, 0);
#else
  *(int4*)dst_lds = *(const int4*)src_g;
#endif
}

__device__ __forceinline__ void zero16(__bf16* dst_lds) {
  *(int4*)dst_lds = make_int4(0, 0, 0, 0);
}

__device__ __forceinline__ void wait_async() {
#if ASYNC_LDS
#if __has_builtin(__builtin_amdgcn_s_wait_asynccnt)
  __builtin_amdgcn_s_wait_asynccnt(0);
#else
  asm volatile("s_wait_asynccnt 0" ::: "memory");
#endif
#endif
}

// ---------------------------------------------------------------------------
// WMMA fragment helpers (wave32, v_wmma_f32_16x16x32_bf16)
// A 16x32 (ISA 7.12.2): lane&15 = M; VGPR v holds K=(v&4?16:0)+8*half+2*(v&3).
// B mirrored (lane&15 = N) from an LDS [col][K] chunk. C/D: lane&15 = N,
// acc[j] = row j + 8*half.
// ---------------------------------------------------------------------------
__device__ __forceinline__ int wmma_kbase(int v, int half) {
  return ((v & 4) ? 16 : 0) + half * 8 + (v & 3) * 2;
}

template<int LD>
__device__ __forceinline__ v16bf load_frag(const __bf16* buf, int rc, int k0, int half) {
  v16bf f;
#pragma unroll
  for (int v = 0; v < 8; ++v) {
    int k = k0 + wmma_kbase(v, half);
    f[2 * v]     = buf[rc * LD + k];
    f[2 * v + 1] = buf[rc * LD + k + 1];
  }
  return f;
}

__device__ __forceinline__ v8f wmma_bf16(v16bf a, v16bf b, v8f c) {
  return __builtin_amdgcn_wmma_f32_16x16x32_bf16(false, a, false, b, (short)0, c,
                                                 false, false);
}

// ---------------------------------------------------------------------------
// Weight prep: [K x N] f32 -> transposed, zero-padded [N x Kpad] bf16.
// ---------------------------------------------------------------------------
__global__ void cvt_transpose(const float* __restrict__ src, __bf16* __restrict__ dst,
                              int K, int N, int Kpad) {
  int e = blockIdx.x * 256 + threadIdx.x;
  if (e >= N * Kpad) return;
  int n = e / Kpad, k = e - n * Kpad;
  dst[e] = (k < K) ? (__bf16)src[(size_t)k * N + n] : (__bf16)0.f;
}

// ---------------------------------------------------------------------------
// Farthest point sampling: one block per cloud, cloud + dmin in LDS.
// ---------------------------------------------------------------------------
__global__ __launch_bounds__(256) void fps_kernel(const float* __restrict__ pos,
                                                  int Np, int ns,
                                                  int* __restrict__ idx_out,
                                                  float* __restrict__ q_out) {
  __shared__ float px[2048], py[2048], pz[2048], dmin[2048];
  __shared__ float rv[256];
  __shared__ int   ri[256];
  const int b = blockIdx.x, tid = threadIdx.x;
  const float* pb = pos + (size_t)b * Np * 3;
  for (int i = tid; i < Np; i += 256) {
    px[i] = pb[i * 3 + 0]; py[i] = pb[i * 3 + 1]; pz[i] = pb[i * 3 + 2];
    dmin[i] = 1e10f;
  }
  if (tid == 0) {
    idx_out[(size_t)b * ns] = 0;
    q_out[(size_t)b * ns * 3 + 0] = pb[0];
    q_out[(size_t)b * ns * 3 + 1] = pb[1];
    q_out[(size_t)b * ns * 3 + 2] = pb[2];
  }
  __syncthreads();
  int last = 0;
  for (int s = 1; s < ns; ++s) {
    const float lx = px[last], ly = py[last], lz = pz[last];
    float best = -1.f; int bi = 0x7fffffff;
    for (int i = tid; i < Np; i += 256) {
      float dx = px[i] - lx, dy = py[i] - ly, dz = pz[i] - lz;
      float d = dx * dx + dy * dy + dz * dz;
      float dm = fminf(dmin[i], d);
      dmin[i] = dm;
      if (dm > best) { best = dm; bi = i; }   // ascending i -> ties keep lowest
    }
    rv[tid] = best; ri[tid] = bi;
    __syncthreads();
    for (int off = 128; off > 0; off >>= 1) {
      if (tid < off) {
        float ov = rv[tid + off]; int oi = ri[tid + off];
        if (ov > rv[tid] || (ov == rv[tid] && oi < ri[tid])) { rv[tid] = ov; ri[tid] = oi; }
      }
      __syncthreads();
    }
    last = ri[0];
    if (tid == 0) {
      idx_out[(size_t)b * ns + s] = last;
      q_out[((size_t)b * ns + s) * 3 + 0] = px[last];
      q_out[((size_t)b * ns + s) * 3 + 1] = py[last];
      q_out[((size_t)b * ns + s) * 3 + 2] = pz[last];
    }
    __syncthreads();
  }
}

// ---------------------------------------------------------------------------
// Radius query: first K=64 in-radius points in index order (pyg semantics).
// ---------------------------------------------------------------------------
__global__ void radius_kernel(const float* __restrict__ pos, const float* __restrict__ q,
                              int Np, int Nq, int total,
                              int* __restrict__ nidx, int* __restrict__ ncnt) {
  int t = blockIdx.x * blockDim.x + threadIdx.x;
  if (t >= total) return;
  int b = t / Nq;
  const float* pb = pos + (size_t)b * Np * 3;
  float qx = q[(size_t)t * 3 + 0], qy = q[(size_t)t * 3 + 1], qz = q[(size_t)t * 3 + 2];
  int* out = nidx + (size_t)t * 64;
  int cnt = 0;
  for (int i = 0; i < Np && cnt < 64; ++i) {
    float dx = pb[i * 3 + 0] - qx, dy = pb[i * 3 + 1] - qy, dz = pb[i * 3 + 2] - qz;
    if (dx * dx + dy * dy + dz * dz <= 0.25f) out[cnt++] = i;
  }
  ncnt[t] = cnt;
  for (int c = cnt; c < 64; ++c) out[c] = 0;
}

// ---------------------------------------------------------------------------
// One MLP layer of the fused SA kernel. Asrc: LDS [64][KD] bf16.
// WgT: global, transposed+padded [Cout][KD] bf16 -> contiguous 16B staging.
// ---------------------------------------------------------------------------
template<int KD>
__device__ __forceinline__ void sa_layer(const __bf16* Asrc, __bf16* Wt,
                                         const __bf16* __restrict__ WgT,
                                         const float* __restrict__ bias, int Cout,
                                         __bf16* Adst, int dstStride, bool relu,
                                         float* red, int redStride, int cnt,
                                         bool do_reduce) {
  const int tid = threadIdx.x;
  const int lane = tid & 31, wv = tid >> 5;
  const int sel = lane & 15, half = lane >> 4;
  constexpr int CPR = KD / 8;            // 16B chunks per row
  for (int ng = 0; ng < Cout; ng += 32) {
    for (int ch = tid; ch < 32 * CPR; ch += 128) {   // stage W chunk [32][KD]
      int c = ch / CPR, kk = (ch - c * CPR) * 8;
      copy16(&Wt[c * KD + kk], &WgT[(size_t)(ng + c) * KD + kk]);
    }
    wait_async();
    __syncthreads();
#pragma unroll
    for (int nt = 0; nt < 2; ++nt) {
      v8f acc = {0.f, 0.f, 0.f, 0.f, 0.f, 0.f, 0.f, 0.f};
#pragma unroll
      for (int kt = 0; kt < KD; kt += 32) {
        v16bf a = load_frag<KD>(Asrc, wv * 16 + sel, kt, half);
        v16bf b = load_frag<KD>(Wt,   nt * 16 + sel, kt, half);
        acc = wmma_bf16(a, b, acc);
      }
      const int col = ng + nt * 16 + sel;
      const float bc = bias[col];
      if (!do_reduce) {
#pragma unroll
        for (int j = 0; j < 8; ++j) {
          int row = wv * 16 + j + 8 * half;
          float v = acc[j] + bc;
          if (relu) v = fmaxf(v, 0.f);
          Adst[row * dstStride + col] = (__bf16)v;
        }
      } else {
        float m = NEGV;
#pragma unroll
        for (int j = 0; j < 8; ++j) {
          int slot = wv * 16 + j + 8 * half;
          float v = (slot < cnt) ? (acc[j] + bc) : NEGV;
          m = fmaxf(m, v);
        }
        m = fmaxf(m, __shfl_xor(m, 16, 32));
        if (half == 0) red[wv * redStride + col] = m;
      }
    }
    __syncthreads();
  }
}

// ---------------------------------------------------------------------------
// Fused set-abstraction conv: one workgroup (4 wave32) per centroid.
// ---------------------------------------------------------------------------
template<int KP, int CIN, int C1, int C2, int C3>
__global__ __launch_bounds__(128) void sa_conv(const float* __restrict__ pos,
                                               const __bf16* __restrict__ xfeat,
                                               const float* __restrict__ q,
                                               const int* __restrict__ nidx,
                                               const int* __restrict__ ncnt,
                                               const __bf16* w1, const __bf16* w2,
                                               const __bf16* w3, const float* b1,
                                               const float* b2, const float* b3,
                                               __bf16* __restrict__ xout,
                                               int Np, int Nq) {
  constexpr int M1 = (KP > C1) ? KP : C1;
  constexpr int AK = (M1 > C2) ? M1 : C2;
  __shared__ alignas(16) __bf16 A0[64 * AK];
  __shared__ alignas(16) __bf16 A1[64 * AK];
  __shared__ alignas(16) __bf16 Wt[32 * AK];
  __shared__ float  red[4 * C3];
  __shared__ int    nloc[64];
  __shared__ float  relloc[64 * 3];

  const int cid = blockIdx.x;
  const int b = cid / Nq, ci = cid - b * Nq;
  const int tid = threadIdx.x;
  const float* posb = pos + (size_t)b * Np * 3;
  const __bf16* xb = (CIN > 3) ? (xfeat + (size_t)b * Np * (CIN - 3)) : nullptr;
  const size_t qi = (size_t)b * Nq + ci;
  const float qx = q[qi * 3 + 0], qy = q[qi * 3 + 1], qz = q[qi * 3 + 2];
  const int cnt = ncnt[qi];

  if (tid < 64) {
    int n = nidx[qi * 64 + tid];
    nloc[tid] = n;
    relloc[tid * 3 + 0] = posb[n * 3 + 0] - qx;
    relloc[tid * 3 + 1] = posb[n * 3 + 1] - qy;
    relloc[tid * 3 + 2] = posb[n * 3 + 2] - qz;
  }
  __syncthreads();
  for (int e = tid; e < 64 * KP; e += 128) {          // build A0 [64][KP]
    int kk = e / KP, ch = e - kk * KP;
    float v = 0.f;
    if (ch < CIN) {
      if (CIN > 3 && ch < CIN - 3) v = (float)xb[(size_t)nloc[kk] * (CIN - 3) + ch];
      else                         v = relloc[kk * 3 + (ch - (CIN - 3))];
    }
    A0[kk * KP + ch] = (__bf16)v;
  }
  __syncthreads();
  sa_layer<KP>(A0, Wt, w1, b1, C1, A1, C1, true, nullptr, 0, 0, false);
  sa_layer<C1>(A1, Wt, w2, b2, C2, A0, C2, true, nullptr, 0, 0, false);
  sa_layer<C2>(A0, Wt, w3, b3, C3, nullptr, 0, false, red, C3, cnt, true);
  __syncthreads();
  for (int c = tid; c < C3; c += 128) {
    float m = fmaxf(fmaxf(red[c], red[C3 + c]), fmaxf(red[2 * C3 + c], red[3 * C3 + c]));
    xout[qi * C3 + c] = (__bf16)m;
  }
}

// ---------------------------------------------------------------------------
// Streaming WMMA GEMM: out = act(A[MxK] * W + bias), W given transposed [N][K].
// Requires K % 32 == 0 and 16B-aligned rows (guaranteed by padding).
// flags: bit0 = ReLU, bit1 = f32 output (else bf16)
// ---------------------------------------------------------------------------
__global__ __launch_bounds__(128) void gemm_bias_act(const __bf16* __restrict__ A,
                                                     const __bf16* __restrict__ WT,
                                                     const float* __restrict__ bias,
                                                     void* outp, int M, int K, int N,
                                                     int lda, int ldo, int flags) {
  __shared__ alignas(16) __bf16 As[64 * 32];      // [row][k]
  __shared__ alignas(16) __bf16 Ws[64 * 32];      // [col][k]
  const int tid = threadIdx.x, lane = tid & 31, wv = tid >> 5;
  const int sel = lane & 15, half = lane >> 4;
  const int row0 = blockIdx.y * 64, col0 = blockIdx.x * 64;
  v8f acc[4];
#pragma unroll
  for (int nt = 0; nt < 4; ++nt) acc[nt] = (v8f){0.f,0.f,0.f,0.f,0.f,0.f,0.f,0.f};
  const int KT = K >> 5;
  for (int kt = 0; kt < KT; ++kt) {
    const int k0 = kt << 5;
    if (tid == 0 && kt + 1 < KT) {                     // -> global_prefetch_b8
      __builtin_prefetch(A + (size_t)row0 * lda + (size_t)(k0 + 32), 0, 1);
      __builtin_prefetch(WT + (size_t)col0 * K + (size_t)(k0 + 32), 0, 1);
    }
    for (int ch = tid; ch < 256; ch += 128) {          // 16B chunks: 64 rows x 4
      int r = ch >> 2, kk = (ch & 3) << 3;
      int gr = row0 + r, gc = col0 + r;
      __bf16* da = &As[r * 32 + kk];
      __bf16* dw = &Ws[r * 32 + kk];
      if (gr < M) copy16(da, A + (size_t)gr * lda + (k0 + kk)); else zero16(da);
      if (gc < N) copy16(dw, WT + (size_t)gc * K + (k0 + kk)); else zero16(dw);
    }
    wait_async();
    __syncthreads();
    v16bf a = load_frag<32>(As, wv * 16 + sel, 0, half);
#pragma unroll
    for (int nt = 0; nt < 4; ++nt) {
      v16bf b = load_frag<32>(Ws, nt * 16 + sel, 0, half);
      acc[nt] = wmma_bf16(a, b, acc[nt]);
    }
    __syncthreads();
  }
  const bool relu = flags & 1, f32o = flags & 2;
#pragma unroll
  for (int nt = 0; nt < 4; ++nt) {
    int col = col0 + nt * 16 + sel;
    if (col >= N) continue;
    float bc = bias ? bias[col] : 0.f;
#pragma unroll
    for (int j = 0; j < 8; ++j) {
      int row = row0 + wv * 16 + j + 8 * half;
      if (row >= M) continue;
      float v = acc[nt][j] + bc;
      if (relu) v = fmaxf(v, 0.f);
      if (f32o) ((float*)outp)[(size_t)row * ldo + col] = v;
      else      ((__bf16*)outp)[(size_t)row * ldo + col] = (__bf16)v;
    }
  }
}

// ---------------------------------------------------------------------------
// Build glob input: concat(x2[.,256], p2[.,3]) -> bf16 [rows x 288] (zero pad)
// ---------------------------------------------------------------------------
__global__ void build_glob(const __bf16* __restrict__ x2, const float* __restrict__ p2,
                           __bf16* __restrict__ Ag, int rows) {
  int e = blockIdx.x * 256 + threadIdx.x;
  if (e >= rows * 288) return;
  int r = e / 288, k = e - r * 288;
  float v = 0.f;
  if (k < 256)      v = (float)x2[(size_t)r * 256 + k];
  else if (k < 259) v = p2[(size_t)r * 3 + (k - 256)];
  Ag[e] = (__bf16)v;
}

// ---------------------------------------------------------------------------
// Global max-pool over rows of each cloud.
// ---------------------------------------------------------------------------
__global__ void maxpool_glob(const __bf16* __restrict__ h, __bf16* __restrict__ g,
                             int rowsPerCloud, int C) {
  int col = blockIdx.x * blockDim.x + threadIdx.x;
  int b = blockIdx.y;
  if (col >= C) return;
  const __bf16* base = h + (size_t)b * rowsPerCloud * C + col;
  float m = NEGV;
  for (int r = 0; r < rowsPerCloud; ++r) m = fmaxf(m, (float)base[(size_t)r * C]);
  g[(size_t)b * C + col] = (__bf16)m;
}

// ---------------------------------------------------------------------------
// Host orchestration
// ---------------------------------------------------------------------------
extern "C" void kernel_launch(void* const* d_in, const int* in_sizes, int n_in,
                              void* d_out, int out_size, void* d_ws, size_t ws_size,
                              hipStream_t stream) {
  (void)in_sizes; (void)n_in; (void)out_size; (void)ws_size;
  const int B = 8, N0 = 2048, N1 = 1536, N2 = 1152;

  const float* pos = (const float*)d_in[0];
  // per-layer dims: W is [K x N] in memory; we store transposed [N x Kpad] bf16
  static const int KDIM[12] = {3, 64, 64, 131, 128, 128, 259, 256, 512, 1024, 512, 256};
  static const int NDIM[12] = {64, 64, 128, 128, 128, 256, 256, 512, 1024, 512, 256, 9};
  static const int KPAD[12] = {32, 64, 64, 160, 128, 128, 288, 256, 512, 1024, 512, 256};
  const float* Bias[12];
  for (int l = 0; l < 12; ++l) Bias[l] = (const float*)d_in[2 + 2 * l];

  char* ws = (char*)d_ws;
  size_t off = 0;
  auto alloc = [&](size_t bytes) -> void* {
    void* p = ws + off;
    off = (off + bytes + 255) & ~(size_t)255;
    return p;
  };

  __bf16* wbT[12];
  for (int l = 0; l < 12; ++l) wbT[l] = (__bf16*)alloc((size_t)NDIM[l] * KPAD[l] * 2);
  int*    idx1  = (int*)alloc((size_t)B * N1 * 4);
  float*  p1    = (float*)alloc((size_t)B * N1 * 3 * 4);
  int*    idx2  = (int*)alloc((size_t)B * N2 * 4);
  float*  p2    = (float*)alloc((size_t)B * N2 * 3 * 4);
  int*    nidx1 = (int*)alloc((size_t)B * N1 * 64 * 4);
  int*    ncnt1 = (int*)alloc((size_t)B * N1 * 4);
  int*    nidx2 = (int*)alloc((size_t)B * N2 * 64 * 4);
  int*    ncnt2 = (int*)alloc((size_t)B * N2 * 4);
  __bf16* x1    = (__bf16*)alloc((size_t)B * N1 * 128 * 2);
  __bf16* x2    = (__bf16*)alloc((size_t)B * N2 * 256 * 2);
  __bf16* Ag    = (__bf16*)alloc((size_t)B * N2 * 288 * 2);
  __bf16* h1    = (__bf16*)alloc((size_t)B * N2 * 256 * 2);
  __bf16* h2    = (__bf16*)alloc((size_t)B * N2 * 512 * 2);
  __bf16* h3    = (__bf16*)alloc((size_t)B * N2 * 1024 * 2);
  __bf16* g     = (__bf16*)alloc((size_t)B * 1024 * 2);
  __bf16* h4    = (__bf16*)alloc((size_t)B * 512 * 2);
  __bf16* h5    = (__bf16*)alloc((size_t)B * 256 * 2);

  for (int l = 0; l < 12; ++l) {
    int n = NDIM[l] * KPAD[l];
    cvt_transpose<<<(n + 255) / 256, 256, 0, stream>>>(
        (const float*)d_in[1 + 2 * l], wbT[l], KDIM[l], NDIM[l], KPAD[l]);
  }

  // --- sa1 ---
  fps_kernel<<<B, 256, 0, stream>>>(pos, N0, N1, idx1, p1);
  radius_kernel<<<(B * N1 + 255) / 256, 256, 0, stream>>>(pos, p1, N0, N1, B * N1,
                                                          nidx1, ncnt1);
  sa_conv<32, 3, 64, 64, 128><<<B * N1, 128, 0, stream>>>(
      pos, nullptr, p1, nidx1, ncnt1, wbT[0], wbT[1], wbT[2], Bias[0], Bias[1], Bias[2],
      x1, N0, N1);

  // --- sa2 ---
  fps_kernel<<<B, 256, 0, stream>>>(p1, N1, N2, idx2, p2);
  radius_kernel<<<(B * N2 + 255) / 256, 256, 0, stream>>>(p1, p2, N1, N2, B * N2,
                                                          nidx2, ncnt2);
  sa_conv<160, 131, 128, 128, 256><<<B * N2, 128, 0, stream>>>(
      p1, x1, p2, nidx2, ncnt2, wbT[3], wbT[4], wbT[5], Bias[3], Bias[4], Bias[5],
      x2, N1, N2);

  // --- glob MLP + max pool ---
  const int rows = B * N2;  // 9216
  build_glob<<<(rows * 288 + 255) / 256, 256, 0, stream>>>(x2, p2, Ag, rows);
  gemm_bias_act<<<dim3(256 / 64, rows / 64), 128, 0, stream>>>(
      Ag, wbT[6], Bias[6], h1, rows, 288, 256, 288, 256, 1);
  gemm_bias_act<<<dim3(512 / 64, rows / 64), 128, 0, stream>>>(
      h1, wbT[7], Bias[7], h2, rows, 256, 512, 256, 512, 1);
  gemm_bias_act<<<dim3(1024 / 64, rows / 64), 128, 0, stream>>>(
      h2, wbT[8], Bias[8], h3, rows, 512, 1024, 512, 1024, 0);
  maxpool_glob<<<dim3(1024 / 256, B), 256, 0, stream>>>(h3, g, N2, 1024);

  // --- final MLP ---
  gemm_bias_act<<<dim3(512 / 64, 1), 128, 0, stream>>>(
      g, wbT[9], Bias[9], h4, B, 1024, 512, 1024, 512, 1);
  gemm_bias_act<<<dim3(256 / 64, 1), 128, 0, stream>>>(
      h4, wbT[10], Bias[10], h5, B, 512, 256, 512, 256, 1);
  gemm_bias_act<<<dim3(1, 1), 128, 0, stream>>>(
      h5, wbT[11], Bias[11], d_out, B, 256, 9, 256, 9, 2);
}
